// EdgeConv_36679020708001
// MI455X (gfx1250) — matrix operand order repaired and checked by reading, not compiled, not run on hardware
//
#include <hip/hip_runtime.h>

#define B_ 8
#define C_ 64
#define N_ 4096
#define O_ 64
#define K_ 20
#define CH 512
#define NCHUNK (N_ / CH)   // 8
#define ROWS 32            // rows per knn block (two 16-row A tiles)
#define PITCH (CH + 4)     // float pitch, keeps 16B alignment per 16-col segment

typedef _Float16 h8  __attribute__((ext_vector_type(8)));
typedef _Float16 h16 __attribute__((ext_vector_type(16)));
typedef float    vf8 __attribute__((ext_vector_type(8)));

union H16 { h16 v; h8 h[2]; };

// ---- WMMA fragment loaders (row-major f16, row pitch = 64 halves) ----------
// A (16x32, MxK): lane l<16: row=l, K = {kb*32+0..7, kb*32+16..23}
//                 lane l>=16: row=l-16, K = {kb*32+8..15, kb*32+24..31}
__device__ inline h16 load_a_frag(const _Float16* __restrict__ base, int kb, int lane) {
    int row = lane & 15;
    int cb  = kb * 32 + ((lane >> 4) << 3);
    const _Float16* p = base + row * 64 + cb;
    H16 r;
    r.h[0] = *(const h8*)(p);
    r.h[1] = *(const h8*)(p + 16);
    return r.v;
}
// B (32x16, KxN): lane l<16: col=l, K = {kb*32+0..7, kb*32+16..23}
//                 lane l>=16: col=l-16, K = {kb*32+8..15, kb*32+24..31}
__device__ inline h16 load_b_frag(const _Float16* __restrict__ base, int kb, int lane) {
    int col = lane & 15;
    int cb  = kb * 32 + ((lane >> 4) << 3);
    const _Float16* p = base + col * 64 + cb;
    H16 r;
    r.h[0] = *(const h8*)(p);
    r.h[1] = *(const h8*)(p + 16);
    return r.v;
}

__device__ inline vf8 wmma16(h16 a, h16 b, vf8 c) {
    return __builtin_amdgcn_wmma_f32_16x16x32_f16(false, a, false, b, (short)0, c, false, false);
}

// monotone float->u32 key for unsigned max selection
__device__ inline unsigned int mono(float f) {
    unsigned int u = __float_as_uint(f);
    return (u & 0x80000000u) ? ~u : (u | 0x80000000u);
}

__device__ inline unsigned long long wave_max_u64(unsigned long long v) {
#pragma unroll
    for (int m = 16; m >= 1; m >>= 1) {
        int lo = (int)(unsigned int)(v);
        int hi = (int)(unsigned int)(v >> 32);
        unsigned int olo = (unsigned int)__shfl_xor(lo, m, 32);
        unsigned int ohi = (unsigned int)__shfl_xor(hi, m, 32);
        unsigned long long o = ((unsigned long long)ohi << 32) | olo;
        v = (o > v) ? o : v;
    }
    return v;
}

// ---- prep: x (B,C,N) f32 -> xh (B,N,C) f16, async-to-LDS tile transpose ----
__global__ __launch_bounds__(256) void prep_kernel(const float* __restrict__ x,
                                                   _Float16* __restrict__ xh) {
    __shared__ float tile[32][33];
    int id = blockIdx.x;
    int nt = id & 127; id >>= 7;
    int ct = id & 1;   id >>= 1;
    int b  = id;
    int tx = threadIdx.x & 31;
    int ty = threadIdx.x >> 5;
#pragma unroll
    for (int i = 0; i < 4; ++i) {
        int c = ct * 32 + ty + i * 8;
        int n = nt * 32 + tx;
        const float* gp = &x[(b * C_ + c) * (size_t)N_ + n];
        unsigned lds_off = (unsigned)(uintptr_t)&tile[ty + i * 8][tx];
        // CDNA5 async global->LDS copy (ASYNCcnt path)
        asm volatile("global_load_async_to_lds_b32 %0, %1, off"
                     :: "v"(lds_off), "v"(gp) : "memory");
    }
    asm volatile("s_wait_asynccnt 0x0" ::: "memory");
    __syncthreads();
#pragma unroll
    for (int i = 0; i < 4; ++i) {
        int n = nt * 32 + ty + i * 8;
        int c = ct * 32 + tx;
        xh[((size_t)b * N_ + n) * C_ + c] = (_Float16)tile[tx][ty + i * 8];
    }
}

// ---- sq: row squared norms -------------------------------------------------
__global__ __launch_bounds__(256) void sq_kernel(const float* __restrict__ x,
                                                 float* __restrict__ sq) {
    int gid = blockIdx.x * 256 + threadIdx.x;   // B*N threads
    int b = gid >> 12;
    int n = gid & (N_ - 1);
    float acc = 0.f;
#pragma unroll 8
    for (int c = 0; c < C_; ++c) {
        float v = x[(b * C_ + c) * (size_t)N_ + n];
        acc += v * v;
    }
    sq[gid] = acc;
}

// ---- wconv: fold BN, split W into (W1-W2) and W2 as f16 --------------------
__global__ __launch_bounds__(64) void wconv_kernel(const float* __restrict__ W,
                                                   const float* __restrict__ gamma,
                                                   const float* __restrict__ beta,
                                                   const float* __restrict__ rmean,
                                                   const float* __restrict__ rvar,
                                                   _Float16* __restrict__ Wse,
                                                   _Float16* __restrict__ Wnb,
                                                   float* __restrict__ scale,
                                                   float* __restrict__ bias) {
    int o = threadIdx.x;
#pragma unroll 4
    for (int c = 0; c < 64; ++c) {
        float w1 = W[o * 128 + c];
        float w2 = W[o * 128 + 64 + c];
        Wse[o * 64 + c] = (_Float16)(w1 - w2);
        Wnb[o * 64 + c] = (_Float16)w2;
    }
    float sc = gamma[o] * rsqrtf(rvar[o] + 1e-5f);
    scale[o] = sc;
    bias[o]  = beta[o] - rmean[o] * sc;
}

// ---- proj: S = xt (W1-W2)^T, P = xt W2^T via WMMA --------------------------
__global__ __launch_bounds__(32) void proj_kernel(const _Float16* __restrict__ xh,
                                                  const _Float16* __restrict__ Wse,
                                                  const _Float16* __restrict__ Wnb,
                                                  float* __restrict__ S,
                                                  float* __restrict__ P) {
    int b  = blockIdx.x >> 8;
    int i0 = (blockIdx.x & 255) * 16;
    int lane = threadIdx.x;
    const _Float16* abase = xh + ((size_t)b * N_ + i0) * C_;
    h16 a0 = load_a_frag(abase, 0, lane);
    h16 a1 = load_a_frag(abase, 1, lane);
#pragma unroll
    for (int ot = 0; ot < 4; ++ot) {
        const _Float16* ws = Wse + ot * 16 * 64;
        const _Float16* wn = Wnb + ot * 16 * 64;
        h16 bs0 = load_b_frag(ws, 0, lane);
        h16 bs1 = load_b_frag(ws, 1, lane);
        h16 bn0 = load_b_frag(wn, 0, lane);
        h16 bn1 = load_b_frag(wn, 1, lane);
        vf8 cs = {}; vf8 cp = {};
        cs = wmma16(a0, bs0, cs);
        cs = wmma16(a1, bs1, cs);
        cp = wmma16(a0, bn0, cp);
        cp = wmma16(a1, bn1, cp);
        int o     = ot * 16 + (lane & 15);
        int rbase = (lane >> 4) * 8;
#pragma unroll
        for (int v = 0; v < 8; ++v) {
            size_t n = (size_t)b * N_ + i0 + rbase + v;
            S[n * O_ + o] = cs[v];
            P[n * O_ + o] = cp[v];
        }
    }
}

// ---- knn: fused distance-GEMM (32-row tile) + streaming per-row top-20 -----
__global__ __launch_bounds__(256) void knn_kernel(const _Float16* __restrict__ xh,
                                                  const float* __restrict__ sq,
                                                  int* __restrict__ knn_idx) {
    __shared__ float pdb[ROWS][PITCH];                 // keyed scores, one chunk
    __shared__ unsigned long long run[ROWS][K_];       // running top-20 per row
    int b    = blockIdx.x >> 7;                        // N/ROWS = 128 tiles
    int i0   = (blockIdx.x & 127) * ROWS;
    int lane = threadIdx.x & 31;
    int wv   = threadIdx.x >> 5;                       // 0..7 waves

    const _Float16* a0base = xh + ((size_t)b * N_ + i0) * C_;
    const _Float16* a1base = a0base + 16 * C_;
    h16 a00 = load_a_frag(a0base, 0, lane);
    h16 a01 = load_a_frag(a0base, 1, lane);
    h16 a10 = load_a_frag(a1base, 0, lane);
    h16 a11 = load_a_frag(a1base, 1, lane);

    // software-pipelined B fragments: prefetch (chunk 0, tile 0)
    const _Float16* bb = xh + ((size_t)b * N_ + (wv * 4) * 16) * C_;
    h16 b0 = load_b_frag(bb, 0, lane);
    h16 b1 = load_b_frag(bb, 1, lane);

    for (int ch = 0; ch < NCHUNK; ++ch) {
        // ---- compute phase: 32 col-tiles / 8 waves = 4 tiles per wave ----
#pragma unroll
        for (int t = 0; t < 4; ++t) {
            h16 cb0 = b0, cb1 = b1;
            // prefetch next tile (possibly first tile of next chunk; global
            // loads may legally be issued ahead of the selection barrier)
            int nch = ch, nt = t + 1;
            if (nt == 4) { nch = ch + 1; nt = 0; }
            if (nch < NCHUNK) {
                const _Float16* nb =
                    xh + ((size_t)b * N_ + nch * CH + (wv * 4 + nt) * 16) * C_;
                b0 = load_b_frag(nb, 0, lane);
                b1 = load_b_frag(nb, 1, lane);
            }
            vf8 c0 = {}; vf8 c1 = {};
            c0 = wmma16(a00, cb0, c0);
            c0 = wmma16(a01, cb1, c0);
            c1 = wmma16(a10, cb0, c1);
            c1 = wmma16(a11, cb1, c1);
            int jt    = ch * CH + (wv * 4 + t) * 16;
            int j     = jt + (lane & 15);
            float adj = 0.5f * sq[b * N_ + j];
            int jloc  = j - ch * CH;
            int rbase = (lane >> 4) * 8;
#pragma unroll
            for (int v = 0; v < 8; ++v) {
                pdb[rbase + v][jloc]      = c0[v] - adj;
                pdb[16 + rbase + v][jloc] = c1[v] - adj;
            }
        }
        __syncthreads();
        // ---- selection: 4 rows per wave; merge chunk into running top-20 ----
#pragma unroll
        for (int rr = 0; rr < 4; ++rr) {
            int r = wv * 4 + rr;
            unsigned long long pk[17];
            const float4* prow = (const float4*)(&pdb[r][0] + lane * 16);
#pragma unroll
            for (int w = 0; w < 4; ++w) {
                float4 q = prow[w];
                float qs[4] = {q.x, q.y, q.z, q.w};
#pragma unroll
                for (int cc = 0; cc < 4; ++cc) {
                    int jloc = lane * 16 + w * 4 + cc;
                    pk[w * 4 + cc] =
                        ((unsigned long long)mono(qs[cc]) << 32) |
                        (unsigned int)(ch * CH + jloc);
                }
            }
            pk[16] = (ch > 0 && lane < K_) ? run[r][lane] : 0ull;
            for (int tsel = 0; tsel < K_; ++tsel) {
                unsigned long long m = pk[0];
#pragma unroll
                for (int s = 1; s < 17; ++s) m = (pk[s] > m) ? pk[s] : m;
                m = wave_max_u64(m);
                if (lane == 0) run[r][tsel] = m;
#pragma unroll
                for (int s = 0; s < 17; ++s) if (pk[s] == m) pk[s] = 0ull;
            }
        }
        __syncthreads();
    }
    // ---- emit indices (run rows owned by this wave; LDS is in-order/wave) --
#pragma unroll
    for (int rr = 0; rr < 4; ++rr) {
        int r = wv * 4 + rr;
        if (lane < K_)
            knn_idx[((size_t)b * N_ + i0 + r) * K_ + lane] =
                (int)(run[r][lane] & 0xffffffffu);
    }
}

// ---- gather: out[b,o,n] = LReLU(scale*(S + max/min_k P[idx]) + bias) -------
__global__ __launch_bounds__(32) void gather_kernel(const float* __restrict__ S,
                                                    const float* __restrict__ P,
                                                    const int* __restrict__ knn_idx,
                                                    const float* __restrict__ scale,
                                                    const float* __restrict__ bias,
                                                    float* __restrict__ out) {
    int b    = blockIdx.x >> 7;
    int n0   = (blockIdx.x & 127) * 32;
    int lane = threadIdx.x;
    int n    = n0 + lane;
    int idxs[K_];
#pragma unroll
    for (int t = 0; t < K_; ++t) idxs[t] = knn_idx[((size_t)b * N_ + n) * K_ + t];
#pragma unroll
    for (int h = 0; h < 2; ++h) {
        float4 vmax[8], vmin[8];
        {
            const float4* p0 = (const float4*)(P + ((size_t)b * N_ + idxs[0]) * O_ + h * 32);
#pragma unroll
            for (int q = 0; q < 8; ++q) { vmax[q] = p0[q]; vmin[q] = p0[q]; }
        }
#pragma unroll 4
        for (int t = 1; t < K_; ++t) {
            const float4* p = (const float4*)(P + ((size_t)b * N_ + idxs[t]) * O_ + h * 32);
#pragma unroll
            for (int q = 0; q < 8; ++q) {
                float4 v = p[q];
                vmax[q].x = fmaxf(vmax[q].x, v.x); vmin[q].x = fminf(vmin[q].x, v.x);
                vmax[q].y = fmaxf(vmax[q].y, v.y); vmin[q].y = fminf(vmin[q].y, v.y);
                vmax[q].z = fmaxf(vmax[q].z, v.z); vmin[q].z = fminf(vmin[q].z, v.z);
                vmax[q].w = fmaxf(vmax[q].w, v.w); vmin[q].w = fminf(vmin[q].w, v.w);
            }
        }
        const float4* sp = (const float4*)(S + ((size_t)b * N_ + n) * O_ + h * 32);
#pragma unroll
        for (int q = 0; q < 8; ++q) {
            float4 sv = sp[q];
            float svs[4] = {sv.x, sv.y, sv.z, sv.w};
            float mxs[4] = {vmax[q].x, vmax[q].y, vmax[q].z, vmax[q].w};
            float mns[4] = {vmin[q].x, vmin[q].y, vmin[q].z, vmin[q].w};
#pragma unroll
            for (int cc = 0; cc < 4; ++cc) {
                int o    = h * 32 + q * 4 + cc;
                float sc = scale[o];
                float mm = (sc > 0.f) ? mxs[cc] : mns[cc];
                float y  = sc * (svs[cc] + mm) + bias[o];
                y = (y >= 0.f) ? y : 0.2f * y;
                out[((size_t)b * O_ + o) * N_ + n] = y;   // coalesced over lanes
            }
        }
    }
}

// ---- host launch -----------------------------------------------------------
extern "C" void kernel_launch(void* const* d_in, const int* in_sizes, int n_in,
                              void* d_out, int out_size, void* d_ws, size_t ws_size,
                              hipStream_t stream) {
    const float* x     = (const float*)d_in[0];
    const float* W     = (const float*)d_in[1];
    const float* gamma = (const float*)d_in[2];
    const float* beta  = (const float*)d_in[3];
    const float* rmean = (const float*)d_in[4];
    const float* rvar  = (const float*)d_in[5];
    float* out = (float*)d_out;

    char* ws = (char*)d_ws;
    size_t off = 0;
    auto alloc = [&](size_t bytes) -> char* {
        char* p = ws + off;
        off += (bytes + 255) & ~(size_t)255;
        return p;
    };
    _Float16* xh  = (_Float16*)alloc((size_t)B_ * N_ * C_ * sizeof(_Float16)); // 4 MiB
    float*    sq  = (float*)   alloc((size_t)B_ * N_ * sizeof(float));         // 128 KiB
    float*    S   = (float*)   alloc((size_t)B_ * N_ * O_ * sizeof(float));    // 8 MiB
    float*    P   = (float*)   alloc((size_t)B_ * N_ * O_ * sizeof(float));    // 8 MiB
    int*      idx = (int*)     alloc((size_t)B_ * N_ * K_ * sizeof(int));      // 2.5 MiB
    _Float16* Wse = (_Float16*)alloc(64 * 64 * sizeof(_Float16));
    _Float16* Wnb = (_Float16*)alloc(64 * 64 * sizeof(_Float16));
    float*    scl = (float*)   alloc(64 * sizeof(float));
    float*    bia = (float*)   alloc(64 * sizeof(float));

    wconv_kernel<<<1, 64, 0, stream>>>(W, gamma, beta, rmean, rvar, Wse, Wnb, scl, bia);
    prep_kernel<<<B_ * 2 * (N_ / 32), 256, 0, stream>>>(x, xh);
    sq_kernel<<<(B_ * N_) / 256, 256, 0, stream>>>(x, sq);
    proj_kernel<<<B_ * (N_ / 16), 32, 0, stream>>>(xh, Wse, Wnb, S, P);
    knn_kernel<<<B_ * (N_ / 32), 256, 0, stream>>>(xh, sq, idx);
    gather_kernel<<<B_ * (N_ / 32), 32, 0, stream>>>(S, P, idx, scl, bia, out);
}
